// GRUEncoder_35656818491799
// MI455X (gfx1250) — matrix operand order repaired and checked by reading, not compiled, not run on hardware
//
#include <hip/hip_runtime.h>
#include <stdint.h>

#define B_  128
#define S_  512
#define H_  512
#define BS_ (B_ * S_)
#define KC_ 16              // H / 32 k-chunks
#define NT_ 32              // H / 16 n-tiles
#define FRAG_ELEMS (KC_ * NT_ * 32 * 16)   // 262144 bf16 per 512x512 matrix (2^18)

typedef __attribute__((ext_vector_type(16))) __bf16        v16bf;
typedef __attribute__((ext_vector_type(8)))  float          v8f;
typedef __attribute__((ext_vector_type(8)))  unsigned short us8;
typedef __attribute__((ext_vector_type(16))) unsigned short us16;

static __device__ __forceinline__ unsigned short f32_to_bf16(float f) {
  union { float f; unsigned u; } v; v.f = f;
  unsigned u = v.u;
  return (unsigned short)((u + 0x7FFFu + ((u >> 16) & 1u)) >> 16);  // RNE
}

// Load 16 bf16 (32 bytes, contiguous per lane) as a WMMA operand vector.
static __device__ __forceinline__ v16bf load_frag(const unsigned short* p) {
  const us8* vp = (const us8*)p;
  us8 a = vp[0];
  us8 b = vp[1];
  us16 c = __builtin_shufflevector(a, b, 0, 1, 2, 3, 4, 5, 6, 7,
                                         8, 9, 10, 11, 12, 13, 14, 15);
  return __builtin_bit_cast(v16bf, c);
}

static __device__ __forceinline__ float sigmoidf_(float x) {
  return 1.0f / (1.0f + __expf(-x));
}

// ---------------------------------------------------------------------------
// Kernel 1: convert the six 512x512 f32 weight matrices into bf16 WMMA
// B-matrix fragment layout: frag[mat][kc][nt][lane][e], where lane's 16
// elements are W[nt*16 + (lane&15)][kc*32 + (lane>>4)*8 + (e&7) + (e>>3)*16]  (table k order).
// ---------------------------------------------------------------------------
__global__ void __launch_bounds__(256) prep_weights(
    const float* __restrict__ W0, const float* __restrict__ W1,
    const float* __restrict__ W2, const float* __restrict__ W3,
    const float* __restrict__ W4, const float* __restrict__ W5,
    unsigned short* __restrict__ out) {
  int idx = blockIdx.x * 256 + threadIdx.x;     // 0 .. 6*FRAG_ELEMS-1
  int mat = idx >> 18;
  int rfl = idx & (FRAG_ELEMS - 1);
  const float* W = (mat == 0) ? W0 : (mat == 1) ? W1 : (mat == 2) ? W2
                 : (mat == 3) ? W3 : (mat == 4) ? W4 : W5;
  int e    = rfl & 15;
  int lane = (rfl >> 4) & 31;
  int nt   = (rfl >> 9) & 31;
  int kc   = rfl >> 14;
  int n    = lane & 15;
  int half = lane >> 4;
  int row  = nt * 16 + n;                 // output index (W's first dim)
  int col  = kc * 32 + half * 8 + (e & 7) + ((e >> 3) << 4);     // reduction index (W's second dim)
  out[idx] = f32_to_bf16(W[row * H_ + col]);
}

// ---------------------------------------------------------------------------
// Kernel 2: fused embedding gather + the 3 input-side projections
//   xz/xr/xg[row, :] = emb[inputs[row]] @ W^T + b    (row = b*S + s)
// One workgroup (16 waves) per 16-row M-tile; 96 output tiles -> 6 per wave.
// ---------------------------------------------------------------------------
__global__ void __launch_bounds__(512) input_proj(
    const int* __restrict__ inputs, const float* __restrict__ emb,
    const unsigned short* __restrict__ wfragIn,   // Wiz,Wir,Wih fragments
    const float* __restrict__ biz, const float* __restrict__ bir,
    const float* __restrict__ bih,
    float* __restrict__ xz, float* __restrict__ xr, float* __restrict__ xg) {
  __shared__ __align__(32) unsigned short xA[KC_ * 32 * 16];   // 16 KB

  const int tIdx  = threadIdx.x;
  const int mrow0 = blockIdx.x * 16;

  // Gather 16 embedding rows, convert to bf16, store in A-fragment layout.
  {
    int m     = tIdx >> 5;            // 0..15 : row within tile
    int kbase = (tIdx & 31) * 16;     // 16 consecutive h per thread
    int tok   = inputs[mrow0 + m];
    const float* er = emb + (size_t)tok * H_;
#pragma unroll
    for (int j = 0; j < 16; ++j) {
      int k    = kbase + j;
      int kc   = k >> 5;
      int krel = k & 31;
      int hh   = (krel >> 3) & 1;
      int ls   = m + 16 * hh;
      int e    = (krel & 7) + ((krel >> 4) << 3);
      xA[(kc * 32 + ls) * 16 + e] = f32_to_bf16(er[k]);
    }
  }
  __syncthreads();

  const int w    = tIdx >> 5;
  const int lane = tIdx & 31;
  const int n    = lane & 15;
  const int half = lane >> 4;

  v8f acc[6];
  const v8f vzero = {0.f, 0.f, 0.f, 0.f, 0.f, 0.f, 0.f, 0.f};
#pragma unroll
  for (int t = 0; t < 6; ++t) acc[t] = vzero;

  for (int kc = 0; kc < KC_; ++kc) {
    v16bf a = load_frag(&xA[(kc * 32 + lane) * 16]);
#pragma unroll
    for (int t = 0; t < 6; ++t) {
      int tid = w * 6 + t;
      int mat = tid >> 5;       // 0=z, 1=r, 2=g   (uniform per wave)
      int nt  = tid & 31;
      const unsigned short* bp =
          wfragIn + (size_t)mat * FRAG_ELEMS +
          ((size_t)((kc * NT_ + nt) * 32 + lane) << 4);
      v16bf b = load_frag(bp);
      acc[t] = __builtin_amdgcn_wmma_f32_16x16x32_bf16(
          false, a, false, b, (short)0, acc[t], false, false);
    }
  }

#pragma unroll
  for (int t = 0; t < 6; ++t) {
    int tid = w * 6 + t;
    int mat = tid >> 5;
    int nt  = tid & 31;
    int col = nt * 16 + n;
    const float* bias = (mat == 0) ? biz : (mat == 1) ? bir : bih;
    float*       dst  = (mat == 0) ? xz  : (mat == 1) ? xr  : xg;
    float bv = bias[col];
#pragma unroll
    for (int r = 0; r < 8; ++r) {
      int m_ = r + 8 * half;
      dst[(size_t)(mrow0 + m_) * H_ + col] = acc[t][r] + bv;
    }
  }
}

// ---------------------------------------------------------------------------
// Kernel 3: persistent recurrent scan. 8 workgroups, each owns 16 batch rows.
// h kept in LDS (f32 + bf16 A-fragment copy); weights streamed from L2 each
// step as bf16 fragments; 512 steps with two workgroup barriers per step.
// ---------------------------------------------------------------------------
__global__ void __launch_bounds__(512, 1) gru_scan(
    const unsigned short* __restrict__ wfragH,   // Whz,Whr,Whh fragments
    const float* __restrict__ bhz, const float* __restrict__ bhr,
    const float* __restrict__ bhh,
    const float* __restrict__ xz, const float* __restrict__ xr,
    const float* __restrict__ xg,
    float* __restrict__ out_ann, float* __restrict__ out_hlast) {
  __shared__ __align__(32) float          hF[16 * H_];          // 32 KB
  __shared__ __align__(32) unsigned short hA[KC_ * 32 * 16];    // 16 KB
  __shared__ __align__(32) float          zb[16 * H_];          // 32 KB
  __shared__ __align__(32) float          rb[16 * H_];          // 32 KB
  __shared__ __align__(32) float          gb[16 * H_];          // 32 KB

  const int tIdx  = threadIdx.x;
  const int bbase = blockIdx.x * 16;

  for (int i = tIdx; i < 16 * H_; i += 512) hF[i] = 0.0f;
  for (int i = tIdx; i < KC_ * 32 * 16; i += 512) hA[i] = 0;
  __syncthreads();

  const int w    = tIdx >> 5;
  const int lane = tIdx & 31;
  const int n    = lane & 15;
  const int half = lane >> 4;
  const int pm   = tIdx >> 5;          // phase-B row
  const int pk0  = (tIdx & 31) * 16;   // phase-B col base
  const v8f vzero = {0.f, 0.f, 0.f, 0.f, 0.f, 0.f, 0.f, 0.f};

  for (int step = 0; step < S_; ++step) {
    // ---- Phase A: preactivations via WMMA -------------------------------
    v8f acc[6];
#pragma unroll
    for (int t = 0; t < 6; ++t) acc[t] = vzero;

    for (int kc = 0; kc < KC_; ++kc) {
      v16bf a = load_frag(&hA[(kc * 32 + lane) * 16]);
#pragma unroll
      for (int t = 0; t < 6; ++t) {
        int tid = w * 6 + t;
        int mat = tid >> 5;
        int nt  = tid & 31;
        const unsigned short* bp =
            wfragH + (size_t)mat * FRAG_ELEMS +
            ((size_t)((kc * NT_ + nt) * 32 + lane) << 4);
        v16bf b = load_frag(bp);
        acc[t] = __builtin_amdgcn_wmma_f32_16x16x32_bf16(
            false, a, false, b, (short)0, acc[t], false, false);
      }
    }

#pragma unroll
    for (int t = 0; t < 6; ++t) {
      int tid = w * 6 + t;
      int mat = tid >> 5;                 // uniform per wave
      int nt  = tid & 31;
      int col = nt * 16 + n;
      const float* bias = (mat == 0) ? bhz : (mat == 1) ? bhr : bhh;
      float bv = bias[col];
#pragma unroll
      for (int r = 0; r < 8; ++r) {
        int m_ = r + 8 * half;
        size_t gi = ((size_t)(bbase + m_) * S_ + step) * H_ + col;
        float pre = acc[t][r] + bv;
        if (mat == 0)      zb[m_ * H_ + col] = sigmoidf_(pre + xz[gi]);
        else if (mat == 1) rb[m_ * H_ + col] = sigmoidf_(pre + xr[gi]);
        else               gb[m_ * H_ + col] = pre;     // xg, tanh in phase B
      }
    }
    __syncthreads();

    // ---- Phase B: elementwise GRU blend + h update ----------------------
    {
      size_t gbase = ((size_t)(bbase + pm) * S_ + step) * H_;
#pragma unroll
      for (int j = 0; j < 16; ++j) {
        int k = pk0 + j;
        float z = zb[pm * H_ + k];
        float r = rb[pm * H_ + k];
        float g = tanhf(xg[gbase + k] + r * gb[pm * H_ + k]);
        float h = hF[pm * H_ + k];
        float hn = (1.0f - z) * g + z * h;
        hF[pm * H_ + k] = hn;
        int kc   = k >> 5;
        int krel = k & 31;
        int hh   = (krel >> 3) & 1;
        int ls   = pm + 16 * hh;
        int e    = (krel & 7) + ((krel >> 4) << 3);
        hA[(kc * 32 + ls) * 16 + e] = f32_to_bf16(hn);
        out_ann[gbase + k] = hn;
      }
    }
    __syncthreads();
  }

  // Final hidden state
#pragma unroll
  for (int j = 0; j < 16; ++j) {
    int k = pk0 + j;
    out_hlast[(size_t)(bbase + pm) * H_ + k] = hF[pm * H_ + k];
  }
}

// ---------------------------------------------------------------------------
extern "C" void kernel_launch(void* const* d_in, const int* in_sizes, int n_in,
                              void* d_out, int out_size, void* d_ws, size_t ws_size,
                              hipStream_t stream) {
  const int*   inputs = (const int*)d_in[0];
  const float* emb = (const float*)d_in[1];
  const float* Wiz = (const float*)d_in[2];
  const float* biz = (const float*)d_in[3];
  const float* Wir = (const float*)d_in[4];
  const float* bir = (const float*)d_in[5];
  const float* Wih = (const float*)d_in[6];
  const float* bih = (const float*)d_in[7];
  const float* Whz = (const float*)d_in[8];
  const float* bhz = (const float*)d_in[9];
  const float* Whr = (const float*)d_in[10];
  const float* bhr = (const float*)d_in[11];
  const float* Whh = (const float*)d_in[12];
  const float* bhh = (const float*)d_in[13];

  unsigned short* wfrag = (unsigned short*)d_ws;                 // 6 * 512 KB
  float* xz = (float*)((char*)d_ws + (size_t)6 * FRAG_ELEMS * 2);
  float* xr = xz + (size_t)BS_ * H_;
  float* xg = xr + (size_t)BS_ * H_;

  float* out_ann   = (float*)d_out;
  float* out_hlast = out_ann + (size_t)B_ * S_ * H_;

  prep_weights<<<(6 * FRAG_ELEMS) / 256, 256, 0, stream>>>(
      Wiz, Wir, Wih, Whz, Whr, Whh, wfrag);
  input_proj<<<BS_ / 16, 512, 0, stream>>>(
      inputs, emb, wfrag, biz, bir, bih, xz, xr, xg);
  gru_scan<<<B_ / 16, 512, 0, stream>>>(
      wfrag + (size_t)3 * FRAG_ELEMS, bhz, bhr, bhh, xz, xr, xg,
      out_ann, out_hlast);
}
